// LSTM_7576322310589
// MI455X (gfx1250) — compile-verified, matrix operand order checked by reading, not run on hardware
//
#include <hip/hip_runtime.h>
#include <hip/hip_bf16.h>
#include <math.h>

// ---------------------------------------------------------------------------
// LSTM classifier for MI455X (gfx1250), wave32 + WMMA f32_16x16x32_f16.
//   VOCAB=32000, H=512, L=5, B=64, T=512, NUM_LABELS=20
// GEMMs run on v_wmma_f32_16x16x32_f16 (f32 accumulate, f16 storage).
// The serial scan double-buffers the per-step gate pre-activations into LDS
// with gfx1250 async LDS copies (GLOBAL_LOAD_ASYNC_TO_LDS_B128 / ASYNCcnt),
// overlapping the next step's operand fetch with the current step's WMMAs.
// ---------------------------------------------------------------------------

typedef _Float16 v16h __attribute__((ext_vector_type(16)));
typedef _Float16 h8   __attribute__((ext_vector_type(8)));
typedef float    v8f  __attribute__((ext_vector_type(8)));

#define Hd   512
#define G4   2048         // 4*H
#define Tn   512
#define Bn   64
#define Mrows (Bn*Tn)     // 32768

__device__ __forceinline__ v8f wmma16(v16h a, v16h b, v8f c) {
  // D = A(16x32 f16) * B(32x16 f16) + C(16x16 f32)
  return __builtin_amdgcn_wmma_f32_16x16x32_f16(false, a, false, b,
                                                (short)0, c, false, false);
}

// Load a 16x32 f16 WMMA operand fragment from a row-major matrix
// (row stride = ld elements). Per ISA layout: lane l (l<16) holds
// K = 0..7 and 16..23 of row M=l; lane l+16 holds K = 8..15 and 24..31.
__device__ __forceinline__ v16h load_frag_g(const _Float16* p, int row0,
                                            int ld, int kbase, int lane) {
  const int m  = lane & 15;
  const int ko = (lane >> 4) << 3;               // 0 or 8
  const _Float16* base = p + (size_t)(row0 + m) * ld + kbase + ko;
  h8 lo = *(const h8*)(base);                    // K = kbase+ko .. +7
  h8 hi = *(const h8*)(base + 16);               // K = kbase+16+ko .. +7
  v16h r;
#pragma unroll
  for (int i = 0; i < 8; ++i) { r[i] = lo[i]; r[8 + i] = hi[i]; }
  return r;
}

// Async copy of 16 bytes per lane, global -> LDS (gfx1250, ASYNCcnt).
// VDST VGPR carries the per-lane LDS byte offset (low 32 bits of the generic
// pointer to a __shared__ object are the wave-relative LDS address).
__device__ __forceinline__ void async_copy_b128(const _Float16* gsrc,
                                                _Float16* ldst) {
  unsigned lds_off = (unsigned)(size_t)ldst;
  asm volatile("global_load_async_to_lds_b128 %0, %1, off"
               :: "v"(lds_off), "v"(gsrc) : "memory");
}
__device__ __forceinline__ void wait_asynccnt0() {
  asm volatile("s_wait_asynccnt 0x0" ::: "memory");
}

// --------------------------- small elementwise kernels ----------------------

__global__ void f32_to_f16_kernel(const float* __restrict__ src,
                                  _Float16* __restrict__ dst, int n) {
  int i = blockIdx.x * blockDim.x + threadIdx.x;
  if (i < n) dst[i] = (_Float16)src[i];
}

// X[b*T+t][k] = emb[tok[b*T+t]][k]  (f16)
__global__ void embed_kernel(const int* __restrict__ tok,
                             const float* __restrict__ emb,
                             _Float16* __restrict__ X) {
  int i = blockIdx.x * blockDim.x + threadIdx.x;   // over 32768*512
  int row = i >> 9;
  int k   = i & 511;
  int t   = tok[row];
  X[i] = (_Float16)emb[(size_t)t * Hd + k];
}

// --------------------------- gate pre-activation GEMM -----------------------
// XG[m][n] = sum_k X[m][k]*Wih[n][k] + bih[n] + bhh[n]
// M=32768, N=2048, K=512.  WG tile 256x128 (8 waves, 4x2), wave tile 64x64.
__global__ __launch_bounds__(256)
void gates_gemm_kernel(const _Float16* __restrict__ X,
                       const _Float16* __restrict__ W,
                       const float* __restrict__ bih,
                       const float* __restrict__ bhh,
                       _Float16* __restrict__ XG) {
  const int lane = threadIdx.x & 31;
  const int w    = threadIdx.x >> 5;       // 0..7
  const int wm   = w >> 1;                 // 0..3
  const int wn   = w & 1;                  // 0..1
  const int M0   = blockIdx.x * 256 + wm * 64;
  const int N0   = blockIdx.y * 128 + wn * 64;
  const int ln   = lane & 15;

  v8f acc[4][4];
#pragma unroll
  for (int ni = 0; ni < 4; ++ni) {
    int N = N0 + ni * 16 + ln;
    float bv = bih[N] + bhh[N];
#pragma unroll
    for (int mi = 0; mi < 4; ++mi)
#pragma unroll
      for (int r = 0; r < 8; ++r) acc[mi][ni][r] = bv;
  }

  for (int kt = 0; kt < Hd; kt += 32) {
    // prefetch next K-slab (gfx1250 global_prefetch_b8)
    if (kt + 32 < Hd) {
      __builtin_prefetch(X + (size_t)(M0 + ln) * Hd + kt + 32, 0, 1);
      __builtin_prefetch(W + (size_t)(N0 + ln) * Hd + kt + 32, 0, 1);
    }
    v16h af[4], bf[4];
#pragma unroll
    for (int mi = 0; mi < 4; ++mi)
      af[mi] = load_frag_g(X, M0 + mi * 16, Hd, kt, lane);
#pragma unroll
    for (int ni = 0; ni < 4; ++ni)
      bf[ni] = load_frag_g(W, N0 + ni * 16, Hd, kt, lane);
#pragma unroll
    for (int mi = 0; mi < 4; ++mi)
#pragma unroll
      for (int ni = 0; ni < 4; ++ni)
        acc[mi][ni] = wmma16(af[mi], bf[ni], acc[mi][ni]);
  }

  const int mhi = (lane >> 4) << 3;        // C layout: M = mhi + r, N = ln
#pragma unroll
  for (int mi = 0; mi < 4; ++mi)
#pragma unroll
    for (int ni = 0; ni < 4; ++ni) {
      int N = N0 + ni * 16 + ln;
#pragma unroll
      for (int r = 0; r < 8; ++r) {
        int Mr = M0 + mi * 16 + mhi + r;
        XG[(size_t)Mr * G4 + N] = (_Float16)acc[mi][ni][r];
      }
    }
}

// ------------------------------- LSTM scan ----------------------------------
// One workgroup per 16-row batch tile (grid=4). 8 waves; wave w owns hidden
// slice [64w, 64w+64): gate rows {j, 512+j, 1024+j, 1536+j}. h_{t-1} lives in
// LDS (16x512 f16); c stays in registers. Per step each wave runs
// 16 tiles x 16 K-steps = 256 WMMAs; meanwhile the async engine stages the
// NEXT step's XG slab (16x2048 f16 = 64KB) into the other half of a
// double-buffered LDS region.
//
// Dynamic LDS layout (f16 elements):
//   [0          , 16*512)              hbuf   (h_{t-1}, 16KB)
//   [16*512     , 16*512 + 16*2048)    xg buffer 0 (64KB)
//   [+16*2048   , ...       )          xg buffer 1 (64KB)
#define SCAN_LDS_ELEMS (16 * Hd + 2 * 16 * G4)

__global__ __launch_bounds__(256)
void lstm_scan_kernel(const _Float16* __restrict__ XG,
                      const _Float16* __restrict__ Whh,
                      _Float16* __restrict__ Xnext,
                      float* __restrict__ feat) {
  extern __shared__ __align__(16) _Float16 smem[];
  _Float16* hbuf  = smem;                      // 16 x 512
  _Float16* xbase = smem + 16 * Hd;            // two 16 x 2048 buffers

  const int tid  = threadIdx.x;
  const int lane = tid & 31;
  const int w    = tid >> 5;
  const int b0   = blockIdx.x * 16;
  const int ln   = lane & 15;
  const int ko   = (lane >> 4) << 3;       // A/B operand K sub-offset
  const int mhi  = (lane >> 4) << 3;       // C layout row group

  for (int i = tid; i < 16 * Hd; i += 256) hbuf[i] = (_Float16)0.f;

  // stage XG slab for t=0 into buffer 0 (16 rows x 2048 cols, 16B chunks)
  for (int c = tid; c < 4096; c += 256) {
    int row = c >> 8;                  // 0..15
    int n8  = (c & 255) << 3;          // element col, multiple of 8
    async_copy_b128(XG + ((size_t)(b0 + row) * Tn + 0) * G4 + n8,
                    xbase + row * G4 + n8);
  }
  wait_asynccnt0();
  __syncthreads();

  v8f cst[4];
#pragma unroll
  for (int j = 0; j < 4; ++j)
#pragma unroll
    for (int r = 0; r < 8; ++r) cst[j][r] = 0.f;

  for (int t = 0; t < Tn; ++t) {
    const _Float16* cur = xbase + (t & 1) * (16 * G4);
    _Float16*       nxt = xbase + ((t + 1) & 1) * (16 * G4);

    // kick off async staging of step t+1 (overlaps the WMMA loop below)
    if (t + 1 < Tn) {
      for (int c = tid; c < 4096; c += 256) {
        int row = c >> 8;
        int n8  = (c & 255) << 3;
        async_copy_b128(XG + ((size_t)(b0 + row) * Tn + (t + 1)) * G4 + n8,
                        nxt + row * G4 + n8);
      }
    }

    // accumulators init from the staged XG slab (LDS)
    v8f acc[4][4];                                   // [gate][j]
#pragma unroll
    for (int q = 0; q < 4; ++q)
#pragma unroll
      for (int j = 0; j < 4; ++j) {
        int N = q * Hd + (w * 4 + j) * 16 + ln;
#pragma unroll
        for (int r = 0; r < 8; ++r)
          acc[q][j][r] = (float)cur[(mhi + r) * G4 + N];
      }

    for (int kt = 0; kt < Hd; kt += 32) {
      // A fragment (h_{t-1}) from LDS
      v16h a;
      {
        const _Float16* hp = &hbuf[ln * Hd + kt + ko];
        h8 lo = *(const h8*)hp;
        h8 hi = *(const h8*)(hp + 16);
#pragma unroll
        for (int i = 0; i < 8; ++i) { a[i] = lo[i]; a[8 + i] = hi[i]; }
      }
#pragma unroll
      for (int q = 0; q < 4; ++q)
#pragma unroll
        for (int j = 0; j < 4; ++j) {
          int N = q * Hd + (w * 4 + j) * 16 + ln;
          const _Float16* bp = Whh + (size_t)N * Hd + kt + ko;
          h8 lo = *(const h8*)bp;
          h8 hi = *(const h8*)(bp + 16);
          v16h bfr;
#pragma unroll
          for (int i = 0; i < 8; ++i) { bfr[i] = lo[i]; bfr[8 + i] = hi[i]; }
          acc[q][j] = wmma16(a, bfr, acc[q][j]);
        }
    }

    __syncthreads();   // everyone done reading h_{t-1}

#pragma unroll
    for (int j = 0; j < 4; ++j) {
#pragma unroll
      for (int r = 0; r < 8; ++r) {
        float iv = 1.f / (1.f + __expf(-acc[0][j][r]));
        float fv = 1.f / (1.f + __expf(-acc[1][j][r]));
        float gv = tanhf(acc[2][j][r]);
        float ov = 1.f / (1.f + __expf(-acc[3][j][r]));
        float c  = fv * cst[j][r] + iv * gv;
        cst[j][r] = c;
        float h  = ov * tanhf(c);
        int M    = mhi + r;
        int col  = (w * 4 + j) * 16 + ln;
        hbuf[M * Hd + col] = (_Float16)h;
        Xnext[((size_t)(b0 + M) * Tn + t) * Hd + col] = (_Float16)h;
        if (t == Tn - 1) feat[(b0 + M) * Hd + col] = h;
      }
    }
    wait_asynccnt0();  // our t+1 slab issues have landed in LDS
    __syncthreads();   // h_t and all waves' async writes visible
  }
}

// ------------------------------- classifier ---------------------------------
__global__ void classifier_kernel(const float* __restrict__ feat,
                                  const float* __restrict__ Wc,
                                  const float* __restrict__ bc,
                                  float* __restrict__ out) {
  int i = blockIdx.x * blockDim.x + threadIdx.x;   // 64*20
  if (i >= Bn * 20) return;
  int b = i / 20, c = i % 20;
  float s = bc[c];
  for (int k = 0; k < Hd; ++k) s += feat[b * Hd + k] * Wc[c * Hd + k];
  out[i] = s;
}

// ------------------------------- launcher -----------------------------------
extern "C" void kernel_launch(void* const* d_in, const int* in_sizes, int n_in,
                              void* d_out, int out_size, void* d_ws, size_t ws_size,
                              hipStream_t stream) {
  const int*   tok  = (const int*)d_in[0];
  const float* emb  = (const float*)d_in[1];
  const float* Wih  = (const float*)d_in[2];   // (5, 2048, 512)
  const float* Whh  = (const float*)d_in[3];   // (5, 2048, 512)
  const float* bih  = (const float*)d_in[4];   // (5, 2048)
  const float* bhh  = (const float*)d_in[5];
  const float* Wcls = (const float*)d_in[6];   // (20, 512)
  const float* bcls = (const float*)d_in[7];
  float* out = (float*)d_out;

  // workspace layout (bytes); total ~212 MB
  char* ws = (char*)d_ws;
  _Float16* Xa    = (_Float16*)(ws + 0);           //  32 MB
  _Float16* Xb    = (_Float16*)(ws + 33554432);    //  32 MB
  _Float16* XG    = (_Float16*)(ws + 67108864);    // 128 MB
  _Float16* Wih16 = (_Float16*)(ws + 201326592);   //  10 MB
  _Float16* Whh16 = (_Float16*)(ws + 211812352);   //  10 MB
  float*    feat  = (float*)   (ws + 222298112);   // 128 KB

  const int wn = 5 * G4 * Hd;                      // 5,242,880
  f32_to_f16_kernel<<<(wn + 255) / 256, 256, 0, stream>>>(Wih, Wih16, wn);
  f32_to_f16_kernel<<<(wn + 255) / 256, 256, 0, stream>>>(Whh, Whh16, wn);

  const int xn = Mrows * Hd;                       // 16,777,216
  embed_kernel<<<xn / 256, 256, 0, stream>>>(tok, emb, Xa);

  const size_t scan_lds = (size_t)SCAN_LDS_ELEMS * sizeof(_Float16); // 144 KB

  _Float16* Xcur = Xa;
  _Float16* Xnxt = Xb;
  for (int l = 0; l < 5; ++l) {
    gates_gemm_kernel<<<dim3(Mrows / 256, G4 / 128), 256, 0, stream>>>(
        Xcur, Wih16 + (size_t)l * G4 * Hd, bih + l * G4, bhh + l * G4, XG);
    lstm_scan_kernel<<<Bn / 16, 256, scan_lds, stream>>>(
        XG, Whh16 + (size_t)l * G4 * Hd, Xnxt, feat);
    _Float16* t = Xcur; Xcur = Xnxt; Xnxt = t;
  }

  classifier_kernel<<<(Bn * 20 + 255) / 256, 256, 0, stream>>>(feat, Wcls,
                                                               bcls, out);
}
